// GPTQLinear_52561809769198
// MI455X (gfx1250) — matrix operand order repaired and checked by reading, not compiled, not run on hardware
//
#include <hip/hip_runtime.h>

typedef __attribute__((ext_vector_type(16))) __bf16 v16bf;
typedef __attribute__((ext_vector_type(8)))  __bf16 v8bf;
typedef __attribute__((ext_vector_type(8)))  float  v8f;
typedef __attribute__((ext_vector_type(4)))  float  v4f;

namespace {
constexpr int Kdim = 4096;   // reduction dim
constexpr int Ndim = 11008;  // output cols
constexpr int Tdim = 8192;   // output rows
constexpr int GRP  = 128;    // GPTQ group size
constexpr int BM = 128, BN = 128, BK = 32;
constexpr int LDSW = BK + 8; // halves per LDS row (pad: 80B stride kills bank conflicts)
}

__global__ __launch_bounds__(256, 2) void gptq_wmma_kernel(
    const float* __restrict__ x,    // [T, K] f32
    const int*   __restrict__ qw,   // [K/8, N] nibbles packed along K
    const int*   __restrict__ qz,   // [K/128, N/8] nibbles packed along N
    const float* __restrict__ sc,   // [K/128, N] f32
    float* __restrict__ out)        // [T, N] f32
{
  __shared__ __bf16 lA[BM * LDSW];  // x tile, bf16, [m][k]
  __shared__ __bf16 lB[BN * LDSW];  // W tile, bf16, [n][k] (k contiguous)

  const int tid  = threadIdx.x;
  const int lane = tid & 31;
  const int wave = tid >> 5;         // 0..7
  const int wm   = (wave >> 2) * 64; // wave m offset within block tile
  const int wn   = (wave & 3) * 32;  // wave n offset within block tile

  const int m0 = blockIdx.y * BM;
  const int n0 = blockIdx.x * BN;

  // ---- A staging: 128x32 f32 = 256 threads x 16 floats ----
  const int arow = tid >> 1;
  const int acol = (tid & 1) * 16;
  const float* aptr = x + (size_t)(m0 + arow) * Kdim + acol;

  // ---- B staging: 4x128 dwords of qweight; thread does kr={h, h+2}, one column ----
  const int bcol = tid & 127;        // n within tile
  const int bkr  = tid >> 7;         // 0..1
  const int gn   = n0 + bcol;
  const int zsh  = (gn & 7) * 4;

  v4f a0, a1, a2, a3;
  int q0, q1;
  float zf, sf;

  auto stage = [&](int k0) {
    a0 = *(const v4f*)(aptr + k0);
    a1 = *(const v4f*)(aptr + k0 + 4);
    a2 = *(const v4f*)(aptr + k0 + 8);
    a3 = *(const v4f*)(aptr + k0 + 12);
    const int kp = k0 >> 3;
    q0 = qw[(size_t)(kp + bkr)     * Ndim + gn];
    q1 = qw[(size_t)(kp + bkr + 2) * Ndim + gn];
    const int g = k0 / GRP;
    zf = (float)(((qz[(size_t)g * (Ndim / 8) + (gn >> 3)] >> zsh) & 15) + 1);
    sf = sc[(size_t)g * Ndim + gn];
  };

  auto commit = [&]() {
    v8bf p0, p1;
#pragma unroll
    for (int i = 0; i < 4; ++i) {
      p0[i]     = (__bf16)a0[i];
      p0[i + 4] = (__bf16)a1[i];
      p1[i]     = (__bf16)a2[i];
      p1[i + 4] = (__bf16)a3[i];
    }
    *(v8bf*)(&lA[arow * LDSW + acol])     = p0;
    *(v8bf*)(&lA[arow * LDSW + acol + 8]) = p1;

    v8bf w0, w1;
#pragma unroll
    for (int j = 0; j < 8; ++j) {
      w0[j] = (__bf16)(((float)((q0 >> (4 * j)) & 15) - zf) * sf);
      w1[j] = (__bf16)(((float)((q1 >> (4 * j)) & 15) - zf) * sf);
    }
    *(v8bf*)(&lB[bcol * LDSW + bkr * 8])       = w0;
    *(v8bf*)(&lB[bcol * LDSW + (bkr + 2) * 8]) = w1;
  };

  const v8f vzero = {0.f, 0.f, 0.f, 0.f, 0.f, 0.f, 0.f, 0.f};
  v8f acc[4][2];
#pragma unroll
  for (int mt = 0; mt < 4; ++mt)
#pragma unroll
    for (int nt = 0; nt < 2; ++nt) acc[mt][nt] = vzero;

  const int lh = lane >> 4;   // half-wave select
  const int ll = lane & 15;

  stage(0);
  for (int k0 = 0; k0 < Kdim; k0 += BK) {
    __syncthreads();
    commit();
    __syncthreads();
    if (k0 + BK < Kdim) stage(k0 + BK);  // overlap next global loads with math

    // ---- hoist ALL fragment loads so the 12 ds_load_b128 issue back-to-back
    //      and the 8 WMMAs run as one dense burst behind a single ds wait ----

    // B fragments: VGPR0-3 = K=lh*8..+7, VGPR4-7 = K=16+lh*8..+7 (matches A)
    v16bf bf[2];
#pragma unroll
    for (int nt = 0; nt < 2; ++nt) {
      const __bf16* p = &lB[(wn + nt * 16 + ll) * LDSW + lh * 8];
      v8bf lo = *(const v8bf*)p;
      v8bf hi = *(const v8bf*)(p + 16);
      bf[nt] = __builtin_shufflevector(lo, hi, 0, 1, 2, 3, 4, 5, 6, 7,
                                       8, 9, 10, 11, 12, 13, 14, 15);
    }

    // A fragments: VGPR0-3 = K=lh*8..+7, VGPR4-7 = K=16+lh*8..+7
    v16bf af[4];
#pragma unroll
    for (int mt = 0; mt < 4; ++mt) {
      const __bf16* p = &lA[(wm + mt * 16 + ll) * LDSW + lh * 8];
      v8bf lo = *(const v8bf*)p;
      v8bf hi = *(const v8bf*)(p + 16);
      af[mt] = __builtin_shufflevector(lo, hi, 0, 1, 2, 3, 4, 5, 6, 7,
                                       8, 9, 10, 11, 12, 13, 14, 15);
    }

#pragma unroll
    for (int mt = 0; mt < 4; ++mt) {
#pragma unroll
      for (int nt = 0; nt < 2; ++nt) {
        acc[mt][nt] = __builtin_amdgcn_wmma_f32_16x16x32_bf16(
            false, af[mt], false, bf[nt], (short)0, acc[mt][nt], false, false);
      }
    }
  }

  // C layout: VGPR r -> M = r + (lane>=16 ? 8 : 0), N = lane&15
#pragma unroll
  for (int mt = 0; mt < 4; ++mt) {
#pragma unroll
    for (int nt = 0; nt < 2; ++nt) {
      const int rbase = m0 + wm + mt * 16 + lh * 8;
      const int c     = n0 + wn + nt * 16 + ll;
#pragma unroll
      for (int r = 0; r < 8; ++r)
        out[(size_t)(rbase + r) * Ndim + c] = acc[mt][nt][r];
    }
  }
}

extern "C" void kernel_launch(void* const* d_in, const int* in_sizes, int n_in,
                              void* d_out, int out_size, void* d_ws, size_t ws_size,
                              hipStream_t stream) {
  (void)in_sizes; (void)n_in; (void)out_size; (void)d_ws; (void)ws_size;
  const float* x  = (const float*)d_in[0];
  const int*   qw = (const int*)d_in[1];
  const int*   qz = (const int*)d_in[2];
  const float* sc = (const float*)d_in[3];
  float* out = (float*)d_out;

  dim3 grid(Ndim / BN, Tdim / BM);  // 86 x 64
  dim3 block(256);
  gptq_wmma_kernel<<<grid, block, 0, stream>>>(x, qw, qz, sc, out);
}
